// MultiHeadSelfAttention_90082644066566
// MI455X (gfx1250) — compile-verified, matrix operand order checked by reading, not run on hardware
//
#include <hip/hip_runtime.h>
#include <hip/hip_bf16.h>
#include <stdint.h>

// ---------------------------------------------------------------------------
// Problem constants (fixed by the reference)
// ---------------------------------------------------------------------------
#define BATCH 4
#define SEQ   2048
#define DIM   1024
#define HEADS 16
#define DEPTH 64          // DIM / HEADS
#define ROWS  (BATCH * SEQ)   // 8192

#define NEG_INF (-__builtin_inff())

typedef __attribute__((ext_vector_type(16))) __bf16 bf16x16;
typedef __attribute__((ext_vector_type(8)))  __bf16 bf16x8;
typedef __attribute__((ext_vector_type(8)))  float  f32x8;

// Build a 16x32 (A) or 32x16 (B) bf16 WMMA fragment from two 16-byte chunks.
// Per the CDNA5 ISA 16-bit operand layout, lane (m|n = lane&15, half = lane>>4)
// holds k = 8*half + [0,8)  in VGPRs 0..3 and k = 16 + 8*half + [0,8) in 4..7.
__device__ __forceinline__ bf16x16 load_frag(const __bf16* p0, const __bf16* p1) {
    union { bf16x16 v; bf16x8 h[2]; } f;
    f.h[0] = *(const bf16x8*)p0;
    f.h[1] = *(const bf16x8*)p1;
    return f.v;
}

__device__ __forceinline__ f32x8 wmma_bf16(bf16x16 a, bf16x16 b, f32x8 c) {
    return __builtin_amdgcn_wmma_f32_16x16x32_bf16(
        /*neg_a=*/false, a, /*neg_b=*/false, b,
        /*c_mod=*/(short)0, c, /*reuse_a=*/false, /*reuse_b=*/false);
}

// CDNA5 async DMA: global -> LDS, 16 bytes per lane, tracked by ASYNCcnt.
// Low 32 bits of a generic pointer to __shared__ are the LDS byte offset
// (LDS aperture occupies ADDR[63:32], per ISA 10.2 aperture mapping).
__device__ __forceinline__ void async_ld_b128(void* lds_ptr, const void* gptr) {
    unsigned lds_off = (unsigned)(uintptr_t)lds_ptr;
    asm volatile("global_load_async_to_lds_b128 %0, %1, off"
                 :: "v"(lds_off), "v"(gptr) : "memory");
}
__device__ __forceinline__ void wait_async0() {
    asm volatile("s_wait_asynccnt 0x0" ::: "memory");
}

// ---------------------------------------------------------------------------
// fp32 -> bf16 elementwise convert
// ---------------------------------------------------------------------------
__global__ __launch_bounds__(256) void f32_to_bf16_kernel(
    const float* __restrict__ in, __bf16* __restrict__ out, int n) {
    int i = blockIdx.x * 256 + threadIdx.x;
    if (i < n) out[i] = (__bf16)in[i];
}

// ---------------------------------------------------------------------------
// Transpose + convert a DIM x DIM fp32 weight into bf16 W^T (N-major, K cont.)
// so WMMA B-fragments read contiguous 16B chunks.  block (32,8), grid (32,32)
// ---------------------------------------------------------------------------
__global__ __launch_bounds__(256) void transpose_w_kernel(
    const float* __restrict__ W, __bf16* __restrict__ Wt) {
    __shared__ float tile[32][33];
    const int nb = blockIdx.x * 32;
    const int kb = blockIdx.y * 32;
    const int tx = threadIdx.x, ty = threadIdx.y;
#pragma unroll
    for (int j = 0; j < 32; j += 8)
        tile[ty + j][tx] = W[(size_t)(kb + ty + j) * DIM + nb + tx];
    __syncthreads();
#pragma unroll
    for (int j = 0; j < 32; j += 8)
        Wt[(size_t)(nb + ty + j) * DIM + kb + tx] = (__bf16)tile[tx][ty + j];
}

// ---------------------------------------------------------------------------
// WMMA GEMM:  C[M,N] = A[M,K] (bf16, row-major) * Bt[N,K]^T (bf16) + bias[N]
// 256 threads = 8 waves.  Block tile 128(M) x 64(N), K step 32.
// Double-buffered LDS staged via global_load_async_to_lds_b128:
// one barrier per K-step, tile t+1 DMA in flight under tile t's WMMAs.
// All four B fragments are preloaded into distinct VGPRs so the compiler can
// batch the ds_load_b128s and issue WMMAs back-to-back with partial dscnt
// waits instead of a full LDS round-trip per WMMA.
// mode 0: bf16 split-head [B,H,S,DEPTH] out   (QKV projections)
// mode 1: fp32 row-major [M,N] out            (output projection)
// ---------------------------------------------------------------------------
__global__ __launch_bounds__(256) void gemm_bf16_wmma(
    const __bf16* __restrict__ A, const __bf16* __restrict__ Bt,
    const float* __restrict__ bias,
    __bf16* __restrict__ outSplit, float* __restrict__ outF32,
    int M, int N, int K, int mode) {
    __shared__ __align__(16) __bf16 As[2][128][40];   // +8 pad, 80B row stride
    __shared__ __align__(16) __bf16 Bs[2][64][40];

    const int tid  = threadIdx.x;
    const int lane = tid & 31;
    const int wave = tid >> 5;
    const int m16  = lane & 15;
    const int half = lane >> 4;
    const int mbase = blockIdx.y * 128;
    const int nbase = blockIdx.x * 64;

    auto issue_tile = [&](int buf, int k0) {
        // A tile 128x32: 512 16B chunks, 2 per thread (async DMA to LDS)
#pragma unroll
        for (int i = 0; i < 2; ++i) {
            int c = tid * 2 + i;
            int row = c >> 2, coff = (c & 3) * 8;
            async_ld_b128(&As[buf][row][coff],
                          A + (size_t)(mbase + row) * K + k0 + coff);
        }
        // B tile 64x32: 256 chunks, 1 per thread
        int row = tid >> 2, coff = (tid & 3) * 8;
        async_ld_b128(&Bs[buf][row][coff],
                      Bt + (size_t)(nbase + row) * K + k0 + coff);
    };

    f32x8 acc[4] = {};
    const int nk = K / 32;
    issue_tile(0, 0);

    for (int t = 0; t < nk; ++t) {
        wait_async0();          // this wave's DMA for tile t has landed in LDS
        __syncthreads();        // every wave's tile-t chunks visible;
                                // also: everyone is done reading buf (t+1)&1
        if (t + 1 < nk) issue_tile((t + 1) & 1, (t + 1) * 32);

        const int buf = t & 1;
        bf16x16 aF = load_frag(&As[buf][wave * 16 + m16][8 * half],
                               &As[buf][wave * 16 + m16][16 + 8 * half]);
        bf16x16 bF[4];
#pragma unroll
        for (int nt = 0; nt < 4; ++nt)
            bF[nt] = load_frag(&Bs[buf][nt * 16 + m16][8 * half],
                               &Bs[buf][nt * 16 + m16][16 + 8 * half]);
#pragma unroll
        for (int nt = 0; nt < 4; ++nt)
            acc[nt] = wmma_bf16(aF, bF[nt], acc[nt]);
    }

    // epilogue: C/D layout value (m = r + 8*half, n = lane&15)
#pragma unroll
    for (int nt = 0; nt < 4; ++nt) {
        const int gn = nbase + nt * 16 + m16;
        const float bv = bias[gn];
#pragma unroll
        for (int r = 0; r < 8; ++r) {
            const int gm = mbase + wave * 16 + r + 8 * half;
            const float v = acc[nt][r] + bv;
            if (mode == 0) {
                const int b = gm >> 11, s = gm & (SEQ - 1);
                const int h = gn >> 6,  d = gn & (DEPTH - 1);
                outSplit[(((size_t)b * HEADS + h) * SEQ + s) * DEPTH + d] = (__bf16)v;
            } else {
                outF32[(size_t)gm * N + gn] = v;
            }
        }
    }
}

// ---------------------------------------------------------------------------
// Flash attention over one (b,h) and 64 query rows per block.
// 128 threads = 4 waves; wave w owns query rows [16w,16w+16).
// K tile staged by async DMA; V loaded to VGPRs and transposed into LDS.
// Output written merged-head bf16 [B,S,DIM].
// ---------------------------------------------------------------------------
__global__ __launch_bounds__(128) void attn_kernel(
    const __bf16* __restrict__ Q, const __bf16* __restrict__ Kx,
    const __bf16* __restrict__ V, const int* __restrict__ mask,
    __bf16* __restrict__ O) {
    __shared__ __align__(16) __bf16 Ks[64][72];       // [key][depth], 144B rows
    __shared__ __align__(16) __bf16 Vt[64][72];       // [depth][key]
    __shared__ __align__(16) __bf16 Ps[4][16][64];    // per-wave P tile

    const int tid  = threadIdx.x;
    const int lane = tid & 31;
    const int wave = tid >> 5;
    const int m16  = lane & 15;
    const int half = lane >> 4;
    const int bh = blockIdx.y;
    const int b  = bh >> 4;
    const int h  = bh & (HEADS - 1);
    const int qbase = blockIdx.x * 64;
    const size_t kvbase = (size_t)bh * SEQ * DEPTH;

    // Q fragments live in registers for the whole kernel (depth 64 = 2 k-steps)
    const __bf16* qrow = Q + kvbase + (size_t)(qbase + wave * 16 + m16) * DEPTH;
    bf16x16 qf[2];
    qf[0] = load_frag(qrow + 8 * half,      qrow + 16 + 8 * half);
    qf[1] = load_frag(qrow + 32 + 8 * half, qrow + 48 + 8 * half);

    f32x8 oacc[4] = {};
    float mrun[8], lrun[8];
#pragma unroll
    for (int r = 0; r < 8; ++r) { mrun[r] = NEG_INF; lrun[r] = 0.0f; }

    for (int kt = 0; kt < SEQ / 64; ++kt) {
        const int kb = kt * 64;
        // ---- stage K tile via async DMA, V tile via VGPR transpose ----
#pragma unroll
        for (int i = 0; i < 4; ++i) {
            int c = tid * 4 + i;                 // 512 chunks of 8 bf16
            int row = c >> 3, coff = (c & 7) * 8;
            async_ld_b128(&Ks[row][coff],
                          Kx + kvbase + (size_t)(kb + row) * DEPTH + coff);
            bf16x8 vv =
                *(const bf16x8*)(V + kvbase + (size_t)(kb + row) * DEPTH + coff);
#pragma unroll
            for (int j = 0; j < 8; ++j) Vt[coff + j][row] = vv[j];
        }
        if (kt + 1 < SEQ / 64) {                 // global_prefetch_b8 next tile
            __builtin_prefetch(Kx + kvbase + (size_t)(kb + 64 + (tid >> 1)) * DEPTH, 0, 1);
            __builtin_prefetch(V  + kvbase + (size_t)(kb + 64 + (tid >> 1)) * DEPTH, 0, 1);
        }
        wait_async0();
        __syncthreads();

        // ---- S = Q K^T: preload 4 B fragments per depth-step, then WMMA ----
        f32x8 sacc[4] = {};
        {
            bf16x16 bF[4];
#pragma unroll
            for (int nt = 0; nt < 4; ++nt)
                bF[nt] = load_frag(&Ks[nt * 16 + m16][8 * half],
                                   &Ks[nt * 16 + m16][16 + 8 * half]);
#pragma unroll
            for (int nt = 0; nt < 4; ++nt)
                sacc[nt] = wmma_bf16(qf[0], bF[nt], sacc[nt]);
#pragma unroll
            for (int nt = 0; nt < 4; ++nt)
                bF[nt] = load_frag(&Ks[nt * 16 + m16][32 + 8 * half],
                                   &Ks[nt * 16 + m16][48 + 8 * half]);
#pragma unroll
            for (int nt = 0; nt < 4; ++nt)
                sacc[nt] = wmma_bf16(qf[1], bF[nt], sacc[nt]);
        }

        // ---- scale + mask ----
        float p[4][8];
        const int mrow0 = qbase + wave * 16;
#pragma unroll
        for (int nt = 0; nt < 4; ++nt) {
            const int gn = kb + nt * 16 + m16;
#pragma unroll
            for (int r = 0; r < 8; ++r) {
                const int gm = mrow0 + r + 8 * half;
                const int mv = mask[((size_t)b * SEQ + gm) * SEQ + gn];
                p[nt][r] = mv ? sacc[nt][r] * 0.125f : NEG_INF;   // 1/sqrt(64)
            }
        }

        // ---- online softmax: row reductions across the 16-lane half ----
        float alpha[8], rsum[8];
#pragma unroll
        for (int r = 0; r < 8; ++r) {
            float rmax = fmaxf(fmaxf(p[0][r], p[1][r]), fmaxf(p[2][r], p[3][r]));
#pragma unroll
            for (int off = 1; off < 16; off <<= 1)
                rmax = fmaxf(rmax, __shfl_xor(rmax, off, 16));
            const float mn = fmaxf(mrun[r], rmax);
            alpha[r] = (mrun[r] == NEG_INF) ? 0.0f : __expf(mrun[r] - mn);
            mrun[r] = mn;
            rsum[r] = 0.0f;
        }
#pragma unroll
        for (int nt = 0; nt < 4; ++nt)
#pragma unroll
            for (int r = 0; r < 8; ++r) {
                const float e = (mrun[r] == NEG_INF) ? 0.0f
                                                     : __expf(p[nt][r] - mrun[r]);
                p[nt][r] = e;
                rsum[r] += e;
            }
#pragma unroll
        for (int r = 0; r < 8; ++r) {
#pragma unroll
            for (int off = 1; off < 16; off <<= 1)
                rsum[r] += __shfl_xor(rsum[r], off, 16);
            lrun[r] = lrun[r] * alpha[r] + rsum[r];
        }
#pragma unroll
        for (int nt = 0; nt < 4; ++nt)
#pragma unroll
            for (int r = 0; r < 8; ++r) oacc[nt][r] *= alpha[r];

        // ---- round-trip P through LDS to re-layout C -> A fragments ----
#pragma unroll
        for (int nt = 0; nt < 4; ++nt)
#pragma unroll
            for (int r = 0; r < 8; ++r)
                Ps[wave][r + 8 * half][nt * 16 + m16] = (__bf16)p[nt][r];
        __syncthreads();

        // ---- O += P V (2 key steps x 4 depth tiles, B frags preloaded) ----
        const __bf16* prow = &Ps[wave][m16][0];
#pragma unroll
        for (int ks = 0; ks < 2; ++ks) {
            bf16x16 aF = load_frag(prow + ks * 32 + 8 * half,
                                   prow + ks * 32 + 16 + 8 * half);
            bf16x16 bF[4];
#pragma unroll
            for (int nt = 0; nt < 4; ++nt)
                bF[nt] = load_frag(&Vt[nt * 16 + m16][ks * 32 + 8 * half],
                                   &Vt[nt * 16 + m16][ks * 32 + 16 + 8 * half]);
#pragma unroll
            for (int nt = 0; nt < 4; ++nt)
                oacc[nt] = wmma_bf16(aF, bF[nt], oacc[nt]);
        }
        __syncthreads();
    }

    // ---- normalize and write merged-head bf16 [B,S,DIM] ----
#pragma unroll
    for (int r = 0; r < 8; ++r) {
        const float inv = (lrun[r] > 0.0f) ? 1.0f / lrun[r] : 0.0f;
        const int gm = qbase + wave * 16 + r + 8 * half;
#pragma unroll
        for (int nt = 0; nt < 4; ++nt) {
            const int gn = h * DEPTH + nt * 16 + m16;
            O[((size_t)b * SEQ + gm) * DIM + gn] = (__bf16)(oacc[nt][r] * inv);
        }
    }
}

// ---------------------------------------------------------------------------
// Launcher
// ---------------------------------------------------------------------------
extern "C" void kernel_launch(void* const* d_in, const int* in_sizes, int n_in,
                              void* d_out, int out_size, void* d_ws, size_t ws_size,
                              hipStream_t stream) {
    const float* x    = (const float*)d_in[0];
    const int*   mask = (const int*)  d_in[1];
    const float* Wq   = (const float*)d_in[2];
    const float* bq   = (const float*)d_in[3];
    const float* Wk   = (const float*)d_in[4];
    const float* bk   = (const float*)d_in[5];
    const float* Wv   = (const float*)d_in[6];
    const float* bv   = (const float*)d_in[7];
    const float* Wo   = (const float*)d_in[8];
    const float* bo   = (const float*)d_in[9];
    float* out = (float*)d_out;

    // workspace carve-up (bytes)
    char* ws = (char*)d_ws;
    const size_t MB = 1024ull * 1024ull;
    __bf16* xb  = (__bf16*)(ws);                    // 16 MB  x  bf16 [ROWS,DIM]
    __bf16* wqt = (__bf16*)(ws + 16 * MB);          //  2 MB  Wq^T
    __bf16* wkt = (__bf16*)(ws + 18 * MB);
    __bf16* wvt = (__bf16*)(ws + 20 * MB);
    __bf16* wot = (__bf16*)(ws + 22 * MB);
    __bf16* Qb  = (__bf16*)(ws + 24 * MB);          // 16 MB  [B,H,S,DEPTH]
    __bf16* Kb  = (__bf16*)(ws + 40 * MB);
    __bf16* Vb  = (__bf16*)(ws + 56 * MB);
    __bf16* AOb = (__bf16*)(ws + 72 * MB);          // 16 MB  [B,S,DIM]

    const int nElems = ROWS * DIM;                  // 8388608
    f32_to_bf16_kernel<<<nElems / 256, 256, 0, stream>>>(x, xb, nElems);

    dim3 tb(32, 8), tg(DIM / 32, DIM / 32);
    transpose_w_kernel<<<tg, tb, 0, stream>>>(Wq, wqt);
    transpose_w_kernel<<<tg, tb, 0, stream>>>(Wk, wkt);
    transpose_w_kernel<<<tg, tb, 0, stream>>>(Wv, wvt);
    transpose_w_kernel<<<tg, tb, 0, stream>>>(Wo, wot);

    dim3 gg(DIM / 64, ROWS / 128);                  // (16, 64)
    gemm_bf16_wmma<<<gg, 256, 0, stream>>>(xb, wqt, bq, Qb, nullptr,
                                           ROWS, DIM, DIM, /*mode=*/0);
    gemm_bf16_wmma<<<gg, 256, 0, stream>>>(xb, wkt, bk, Kb, nullptr,
                                           ROWS, DIM, DIM, 0);
    gemm_bf16_wmma<<<gg, 256, 0, stream>>>(xb, wvt, bv, Vb, nullptr,
                                           ROWS, DIM, DIM, 0);

    dim3 ag(SEQ / 64, BATCH * HEADS);               // (32, 64)
    attn_kernel<<<ag, 128, 0, stream>>>(Qb, Kb, Vb, mask, AOb);

    gemm_bf16_wmma<<<gg, 256, 0, stream>>>(AOb, wot, bo, nullptr, out,
                                           ROWS, DIM, DIM, /*mode=*/1);
}
